// Q2B_70841190580384
// MI455X (gfx1250) — matrix operand order repaired and checked by reading, not compiled, not run on hardware
//
#include <hip/hip_runtime.h>
#include <cstdint>

typedef __attribute__((ext_vector_type(16))) _Float16 v16h;
typedef __attribute__((ext_vector_type(8)))  _Float16 v8h;
typedef __attribute__((ext_vector_type(8)))  float    v8f;

#define D_DIM 400
#define KPAD  416          // 13 * 32
#define B_SZ  4096
#define NNEG  64
#define M3    (3 * B_SZ)   // 12288 rows for the 3 query branches
#define GAMMA_C 24.0f
#define CEN_C   0.02f

// ---------------- weight f32 -> f16 [400 x 416] (K padded with zeros) ----------------
__global__ void k_prep_w(const float* __restrict__ W, _Float16* __restrict__ Wh) {
  const int n = blockIdx.x;                       // output-feature row (= B-matrix column)
  for (int k = threadIdx.x; k < KPAD; k += blockDim.x) {
    _Float16 v = (_Float16)0.0f;
    if (k < D_DIM) v = (_Float16)W[(size_t)n * D_DIM + k];
    Wh[(size_t)n * KPAD + k] = v;
  }
}

// ---------------- gather: centers = E[anchor]+R[rel]; offsets = O[rel] ----------------
__global__ void k_gather(const float* __restrict__ ent, const float* __restrict__ rel,
                         const float* __restrict__ offemb,
                         const int* __restrict__ anchor, const int* __restrict__ relid,
                         float* __restrict__ cenF, float* __restrict__ offF,
                         _Float16* __restrict__ cenH, _Float16* __restrict__ offH) {
  const int i = blockIdx.x;                       // row in [0, 3B): q*B + b
  const long a = anchor[i];
  const long r = relid[i];
  const float* ep = ent    + a * D_DIM;
  const float* rp = rel    + r * D_DIM;
  const float* op = offemb + r * D_DIM;
  const size_t rowF = (size_t)i * D_DIM, rowH = (size_t)i * KPAD;
  for (int d = threadIdx.x; d < KPAD; d += blockDim.x) {
    if (d < D_DIM) {
      float c = ep[d] + rp[d];
      float o = op[d];
      cenF[rowF + d] = c;              offF[rowF + d] = o;
      cenH[rowH + d] = (_Float16)c;    offH[rowH + d] = (_Float16)o;
    } else {
      cenH[rowH + d] = (_Float16)0.0f; offH[rowH + d] = (_Float16)0.0f;
    }
  }
}

// ---------------- WMMA GEMM: C[m,n] = act( sum_k A[m,k]*W[n,k] + bias[n] ) ----------------
// One 16x16 tile per wave, 8 waves per block. A: [M, KPAD] f16, Bw: [400, KPAD] f16.
// ACT: 0 = none, 1 = relu, 2 = sigmoid.  OUTH: f16 [.,KPAD] output vs f32 [.,400] output.
template<int ACT, bool OUTH>
__global__ void k_gemm_wmma(const _Float16* __restrict__ A,
                            const _Float16* __restrict__ Bw,
                            const float* __restrict__ bias,
                            _Float16* __restrict__ Ch,
                            float* __restrict__ Cf) {
  const int lane = threadIdx.x & 31;
  const int wave = threadIdx.x >> 5;
  const int m0 = (blockIdx.x * 8 + wave) * 16;
  const int n0 = blockIdx.y * 16;

  // A fragment layout (16-bit A 16x32): lanes 0-15 row M=lane hold K {0..7, 16..23};
  // lanes 16-31 row M=lane-16 hold K {8..15, 24..31}.
  const _Float16* ap = A  + (size_t)(m0 + (lane & 15)) * KPAD;
  const int aLo = (lane >> 4) << 3;                       // 0 or 8
  // B fragment layout (16-bit B 32x16): lanes 0-15 col N=lane hold K {0..7, 16..23};
  // lanes 16-31 col N=lane-16 hold K {8..15, 24..31}. W row n == B column n (contiguous in K).
  const _Float16* bp = Bw + (size_t)(n0 + (lane & 15)) * KPAD + ((lane >> 4) << 3);

  v8f acc = {};
  #pragma unroll
  for (int kt = 0; kt < KPAD / 32; ++kt) {
    const int k0 = kt * 32;
    v8h a0 = *(const v8h*)(ap + k0 + aLo);
    v8h a1 = *(const v8h*)(ap + k0 + 16 + aLo);
    v8h b0 = *(const v8h*)(bp + k0);
    v8h b1 = *(const v8h*)(bp + k0 + 16);
    v16h av = __builtin_shufflevector(a0, a1, 0,1,2,3,4,5,6,7,8,9,10,11,12,13,14,15);
    v16h bv = __builtin_shufflevector(b0, b1, 0,1,2,3,4,5,6,7,8,9,10,11,12,13,14,15);
    acc = __builtin_amdgcn_wmma_f32_16x16x32_f16(false, av, false, bv,
                                                 (short)0, acc, false, false);
  }

  // C/D layout: VGPR r -> lanes 0-15: (M=r, N=lane); lanes 16-31: (M=r+8, N=lane-16)
  const int col   = n0 + (lane & 15);
  const float bb  = bias[col];
  const int rbase = m0 + ((lane >> 4) << 3);
  #pragma unroll
  for (int r = 0; r < 8; ++r) {
    float v = acc[r] + bb;
    if (ACT == 1) v = fmaxf(v, 0.0f);
    if (ACT == 2) v = 1.0f / (1.0f + __expf(-v));
    if (OUTH) Ch[(size_t)(rbase + r) * KPAD  + col] = (_Float16)v;
    else      Cf[(size_t)(rbase + r) * D_DIM + col] = v;
  }
  if (OUTH && blockIdx.y == (D_DIM / 16 - 1)) {
    // zero the K-pad columns 400..415 for this tile's 16 rows
    const int row = m0 + (lane >> 1);
    v8h z = {};
    *(v8h*)(Ch + (size_t)row * KPAD + D_DIM + ((lane & 1) << 3)) = z;
  }
}

// ---------------- mean over the 3 branches of l1, to f16 padded ----------------
__global__ void k_mean(const float* __restrict__ l1, _Float16* __restrict__ mh) {
  const int b = blockIdx.x;
  for (int d = threadIdx.x; d < KPAD; d += blockDim.x) {
    float v = 0.0f;
    if (d < D_DIM) {
      v = (l1[((size_t)0 * B_SZ + b) * D_DIM + d] +
           l1[((size_t)1 * B_SZ + b) * D_DIM + d] +
           l1[((size_t)2 * B_SZ + b) * D_DIM + d]) * (1.0f / 3.0f);
    }
    mh[(size_t)b * KPAD + d] = (_Float16)v;
  }
}

// ---------------- softmax over q, attention-sum center, gated-min offset ----------------
__global__ void k_combine(const float* __restrict__ logits, const float* __restrict__ cenF,
                          const float* __restrict__ offF, const float* __restrict__ gate,
                          float* __restrict__ center, float* __restrict__ offset) {
  const int b = blockIdx.x;
  for (int d = threadIdx.x; d < D_DIM; d += blockDim.x) {
    const size_t i0 = ((size_t)0 * B_SZ + b) * D_DIM + d;
    const size_t i1 = ((size_t)1 * B_SZ + b) * D_DIM + d;
    const size_t i2 = ((size_t)2 * B_SZ + b) * D_DIM + d;
    float l0 = logits[i0], l1v = logits[i1], l2 = logits[i2];
    float mx = fmaxf(l0, fmaxf(l1v, l2));
    float e0 = __expf(l0 - mx), e1 = __expf(l1v - mx), e2 = __expf(l2 - mx);
    float inv = 1.0f / (e0 + e1 + e2);
    float c = (e0 * cenF[i0] + e1 * cenF[i1] + e2 * cenF[i2]) * inv;
    float om = fminf(offF[i0], fminf(offF[i1], offF[i2]));
    center[(size_t)b * D_DIM + d] = c;
    offset[(size_t)b * D_DIM + d] = om * gate[(size_t)b * D_DIM + d];
  }
}

// ---------------- box score: one wave per (b, target) row; out[B, 1+N] ----------------
__global__ void k_score(const float* __restrict__ ent, const int* __restrict__ ans,
                        const int* __restrict__ neg, const float* __restrict__ center,
                        const float* __restrict__ offset, float* __restrict__ out) {
  const int row  = blockIdx.x * 8 + (threadIdx.x >> 5);
  const int lane = threadIdx.x & 31;
  const int b = row / (NNEG + 1);
  const int j = row % (NNEG + 1);
  const long t = (j == 0) ? (long)ans[b] : (long)neg[(size_t)b * NNEG + (j - 1)];
  const float4* tp = (const float4*)(ent + t * D_DIM);
  const float4* cp = (const float4*)(center + (size_t)b * D_DIM);
  const float4* op = (const float4*)(offset + (size_t)b * D_DIM);
  float s_out = 0.0f, s_in = 0.0f;
  for (int i = lane; i < D_DIM / 4; i += 32) {
    float4 tv = tp[i], cv = cp[i], ov = op[i];
    float dx;
    dx = fabsf(tv.x - cv.x); s_out += fmaxf(dx - ov.x, 0.0f); s_in += fminf(dx, ov.x);
    dx = fabsf(tv.y - cv.y); s_out += fmaxf(dx - ov.y, 0.0f); s_in += fminf(dx, ov.y);
    dx = fabsf(tv.z - cv.z); s_out += fmaxf(dx - ov.z, 0.0f); s_in += fminf(dx, ov.z);
    dx = fabsf(tv.w - cv.w); s_out += fmaxf(dx - ov.w, 0.0f); s_in += fminf(dx, ov.w);
  }
  #pragma unroll
  for (int m = 16; m; m >>= 1) {
    s_out += __shfl_xor(s_out, m, 32);
    s_in  += __shfl_xor(s_in,  m, 32);
  }
  if (lane == 0) out[row] = GAMMA_C - s_out - CEN_C * s_in;
}

extern "C" void kernel_launch(void* const* d_in, const int* in_sizes, int n_in,
                              void* d_out, int out_size, void* d_ws, size_t ws_size,
                              hipStream_t stream) {
  (void)in_sizes; (void)n_in; (void)out_size; (void)ws_size;
  const float* ent  = (const float*)d_in[0];
  const float* rel  = (const float*)d_in[1];
  const float* offe = (const float*)d_in[2];
  const float* c_w1 = (const float*)d_in[3];
  const float* c_b1 = (const float*)d_in[4];
  const float* c_w2 = (const float*)d_in[5];
  const float* c_b2 = (const float*)d_in[6];
  const float* o_w1 = (const float*)d_in[7];
  const float* o_b1 = (const float*)d_in[8];
  const float* o_w2 = (const float*)d_in[9];
  const float* o_b2 = (const float*)d_in[10];
  const int* anchor = (const int*)d_in[11];
  const int* relid  = (const int*)d_in[12];
  const int* ans    = (const int*)d_in[13];
  const int* neg    = (const int*)d_in[14];
  float* out = (float*)d_out;

  char* ws = (char*)d_ws;
  size_t off = 0;
  auto take = [&](size_t bytes) -> char* {
    char* p = ws + off;
    off = (off + bytes + 255) & ~(size_t)255;
    return p;
  };
  float*    cenF   = (float*)   take((size_t)M3 * D_DIM * 4);
  float*    offF   = (float*)   take((size_t)M3 * D_DIM * 4);
  _Float16* cenH   = (_Float16*)take((size_t)M3 * KPAD * 2);
  _Float16* offH   = (_Float16*)take((size_t)M3 * KPAD * 2);
  _Float16* wc1h   = (_Float16*)take((size_t)D_DIM * KPAD * 2);
  _Float16* wc2h   = (_Float16*)take((size_t)D_DIM * KPAD * 2);
  _Float16* wo1h   = (_Float16*)take((size_t)D_DIM * KPAD * 2);
  _Float16* wo2h   = (_Float16*)take((size_t)D_DIM * KPAD * 2);
  _Float16* a1h    = (_Float16*)take((size_t)M3 * KPAD * 2);
  float*    logitF = (float*)   take((size_t)M3 * D_DIM * 4);
  float*    l1F    = (float*)   take((size_t)M3 * D_DIM * 4);
  _Float16* ml1h   = (_Float16*)take((size_t)B_SZ * KPAD * 2);
  float*    gateF  = (float*)   take((size_t)B_SZ * D_DIM * 4);
  float*    centC  = (float*)   take((size_t)B_SZ * D_DIM * 4);
  float*    offC   = (float*)   take((size_t)B_SZ * D_DIM * 4);

  // 1) f16 weight staging (W row n == B-matrix column n, K padded to 416)
  k_prep_w<<<D_DIM, 128, 0, stream>>>(c_w1, wc1h);
  k_prep_w<<<D_DIM, 128, 0, stream>>>(c_w2, wc2h);
  k_prep_w<<<D_DIM, 128, 0, stream>>>(o_w1, wo1h);
  k_prep_w<<<D_DIM, 128, 0, stream>>>(o_w2, wo2h);

  // 2) gather branch centers / offsets (f32 + padded f16)
  k_gather<<<M3, 128, 0, stream>>>(ent, rel, offe, anchor, relid,
                                   cenF, offF, cenH, offH);

  // 3) WMMA GEMMs
  dim3 g3(M3 / 128, D_DIM / 16);     // 96 x 25
  dim3 g1(B_SZ / 128, D_DIM / 16);   // 32 x 25
  // a1 = relu(centers @ c_w1^T + c_b1) -> f16
  k_gemm_wmma<1, true ><<<g3, 256, 0, stream>>>(cenH, wc1h, c_b1, a1h, nullptr);
  // logits = a1 @ c_w2^T + c_b2 -> f32
  k_gemm_wmma<0, false><<<g3, 256, 0, stream>>>(a1h, wc2h, c_b2, nullptr, logitF);
  // l1 = relu(offsets @ o_w1^T + o_b1) -> f32
  k_gemm_wmma<1, false><<<g3, 256, 0, stream>>>(offH, wo1h, o_b1, nullptr, l1F);
  // mean over branches -> f16
  k_mean<<<B_SZ, 128, 0, stream>>>(l1F, ml1h);
  // gate = sigmoid(mean_l1 @ o_w2^T + o_b2) -> f32
  k_gemm_wmma<2, false><<<g1, 256, 0, stream>>>(ml1h, wo2h, o_b2, nullptr, gateF);

  // 4) softmax-attention center + gated-min offset
  k_combine<<<B_SZ, 128, 0, stream>>>(logitF, cenF, offF, gateF, centC, offC);

  // 5) box scores: [B, 1+N], one wave per row (HBM-bound gather, the real roofline)
  k_score<<<(B_SZ * (NNEG + 1)) / 8, 256, 0, stream>>>(ent, ans, neg, centC, offC, out);
}
